// GraphSAGE_20547123544332
// MI455X (gfx1250) — matrix operand order repaired and checked by reading, not compiled, not run on hardware
//
#include <hip/hip_runtime.h>
#include <math.h>

// ---------------------------------------------------------------------------
// GraphSAGE forward for MI455X (gfx1250, wave32, WMMA).
// Only layer l=2 of the reference matters (h is overwritten every iteration
// and each layer reads the ORIGINAL x/agg).
// Pipeline: zero ws -> edge scatter (f32 atomics) -> fp16 pre-conversion of
// x / normalized-agg / weights -> WMMA layer+pool kernel (pure f16 loads +
// v_wmma_f32_16x16x32_f16, 32 nodes/block to halve weight re-fetch) ->
// deterministic reduce + MLP head + log_softmax.
// ---------------------------------------------------------------------------

#define N_NODES 100000
#define N_CLS   10
#define D_IN    128
#define D_HID   256
#define NODES_PER_BLOCK 32
#define LAYER_BLOCKS (N_NODES / NODES_PER_BLOCK)   // 3125, exact

typedef __attribute__((ext_vector_type(16))) _Float16 v16h;
typedef __attribute__((ext_vector_type(8)))  _Float16 v8h;
typedef __attribute__((ext_vector_type(4)))  _Float16 v4h;
typedef __attribute__((ext_vector_type(8)))  float    v8f;

// ---------------------------------------------------------------------------
// Stage 0: zero the f32 part of the workspace (deg | partials | aggsum).
// ---------------------------------------------------------------------------
__global__ __launch_bounds__(256) void zero_ws_kernel(float* __restrict__ ws, size_t total) {
    size_t i = (size_t)blockIdx.x * 256 + threadIdx.x;
    if (i < total) ws[i] = 0.0f;
}

// ---------------------------------------------------------------------------
// Stage 1: edge scatter-add.  32 lanes per edge: gather x[src] (4 f32/lane),
// atomic-add into aggsum[tgt]; lane 0 bumps deg[tgt].  x (51 MB) and aggsum
// (51 MB) are L2-resident (192 MB L2).
// ---------------------------------------------------------------------------
__global__ __launch_bounds__(256) void edge_agg_kernel(const int* __restrict__ ei,
                                                       const float* __restrict__ x,
                                                       float* __restrict__ aggsum,
                                                       float* __restrict__ deg,
                                                       int nEdges) {
    int e    = blockIdx.x * 8 + (threadIdx.x >> 5);
    int lane = threadIdx.x & 31;
    if (e >= nEdges) return;
    int src = ei[e];
    int tgt = ei[nEdges + e];
    const float4 v = *reinterpret_cast<const float4*>(x + (size_t)src * D_IN + lane * 4);
    float* dst = aggsum + (size_t)tgt * D_IN + lane * 4;
    __hip_atomic_fetch_add(dst + 0, v.x, __ATOMIC_RELAXED, __HIP_MEMORY_SCOPE_AGENT);
    __hip_atomic_fetch_add(dst + 1, v.y, __ATOMIC_RELAXED, __HIP_MEMORY_SCOPE_AGENT);
    __hip_atomic_fetch_add(dst + 2, v.z, __ATOMIC_RELAXED, __HIP_MEMORY_SCOPE_AGENT);
    __hip_atomic_fetch_add(dst + 3, v.w, __ATOMIC_RELAXED, __HIP_MEMORY_SCOPE_AGENT);
    if (lane == 0)
        __hip_atomic_fetch_add(&deg[tgt], 1.0f, __ATOMIC_RELAXED, __HIP_MEMORY_SCOPE_AGENT);
}

// ---------------------------------------------------------------------------
// fp32 -> fp16 converters (bandwidth-trivial; keeps all v_cvt out of the
// WMMA kernel).  cvt_f16_kernel also converts the weight matrices.
// ---------------------------------------------------------------------------
__global__ __launch_bounds__(256) void cvt_f16_kernel(const float* __restrict__ src,
                                                      _Float16* __restrict__ dst,
                                                      size_t n4) {
    size_t i = (size_t)blockIdx.x * 256 + threadIdx.x;
    if (i >= n4) return;
    float4 v = reinterpret_cast<const float4*>(src)[i];
    v4h h;
    h[0] = (_Float16)v.x; h[1] = (_Float16)v.y; h[2] = (_Float16)v.z; h[3] = (_Float16)v.w;
    reinterpret_cast<v4h*>(dst)[i] = h;
}

// agg mean-normalization folded into the fp16 conversion: agg = aggsum/clip(deg,1)
__global__ __launch_bounds__(256) void cvt_agg_kernel(const float* __restrict__ aggsum,
                                                      const float* __restrict__ deg,
                                                      _Float16* __restrict__ aggh,
                                                      size_t n4) {
    size_t i = (size_t)blockIdx.x * 256 + threadIdx.x;
    if (i >= n4) return;
    size_t node = i / (D_IN / 4);
    float s = 1.0f / fmaxf(deg[node], 1.0f);
    float4 v = reinterpret_cast<const float4*>(aggsum)[i];
    v4h h;
    h[0] = (_Float16)(v.x * s); h[1] = (_Float16)(v.y * s);
    h[2] = (_Float16)(v.z * s); h[3] = (_Float16)(v.w * s);
    reinterpret_cast<v4h*>(aggh)[i] = h;
}

// ---------------------------------------------------------------------------
// WMMA operand loads (pure fp16, 16B-aligned).
// A (16x32 f16, ISA 7.12.2): lane = half*16 + m, row M = m.
//   halves 0..7  <- K = k0 + half*8 + {0..7}
//   halves 8..15 <- K = k0 + 16 + half*8 + {0..7}
// B (32x16 f16): lane holds column n; same K mapping as A.
// ---------------------------------------------------------------------------
__device__ __forceinline__ v16h load_a16(const _Float16* __restrict__ row, int k0, int half) {
    const int ka = k0 + half * 8;
    v8h lo = *reinterpret_cast<const v8h*>(row + ka);
    v8h hi = *reinterpret_cast<const v8h*>(row + ka + 16);
    v16h a;
#pragma unroll
    for (int j = 0; j < 8; ++j) { a[j] = lo[j]; a[j + 8] = hi[j]; }
    return a;
}

__device__ __forceinline__ v16h load_b16(const _Float16* __restrict__ row, int k0, int half) {
    const int kb = k0 + half * 8;
    v8h lo = *reinterpret_cast<const v8h*>(row + kb);
    v8h hi = *reinterpret_cast<const v8h*>(row + kb + 16);
    v16h b;
#pragma unroll
    for (int j = 0; j < 8; ++j) { b[j] = lo[j]; b[j + 8] = hi[j]; }
    return b;
}

__device__ __forceinline__ v8f wmma_f16(v16h a, v16h b, v8f c) {
    return __builtin_amdgcn_wmma_f32_16x16x32_f16(false, a, false, b, (short)0, c,
                                                  false, false);
}

// ---------------------------------------------------------------------------
// Stage 2: h = relu(agg @ Wl2^T + x @ Wr2^T + b2) for 32 nodes per block,
// fused with global mean-pool (per-block column partials, deterministic).
// 8 wave32 per block; wave w owns column tiles {2w, 2w+1} for BOTH node
// tiles (base, base+16): 4 accumulators, 8 v_wmma per K-chunk against
// 16 x 16B f16 loads -> 2x the compute density of the 16-node version and
// half the L2 weight re-fetch (~400 MB total).
// No divergence before WMMA (EXEC all ones; 100000 = 3125*32 exactly).
// ---------------------------------------------------------------------------
__global__ __launch_bounds__(256)
void sage_layer_kernel(const _Float16* __restrict__ xh,
                       const _Float16* __restrict__ aggh,
                       const _Float16* __restrict__ wlh,   // [256,128] f16
                       const _Float16* __restrict__ wrh,   // [256,128] f16
                       const float* __restrict__ bias2,    // [256] f32
                       float* __restrict__ partials)       // [gridDim.x*256]
{
    __shared__ float sPool[D_HID];
    const int tid = threadIdx.x;
    sPool[tid] = 0.0f;
    __syncthreads();

    const int wave = tid >> 5;
    const int lane = tid & 31;
    const int m    = lane & 15;
    const int half = lane >> 4;
    const int node0 = blockIdx.x * NODES_PER_BLOCK + m;   // node tile 0 row
    const int node1 = node0 + 16;                         // node tile 1 row

    const _Float16* xrow0 = xh   + (size_t)node0 * D_IN;
    const _Float16* arow0 = aggh + (size_t)node0 * D_IN;
    const _Float16* xrow1 = xh   + (size_t)node1 * D_IN;
    const _Float16* arow1 = aggh + (size_t)node1 * D_IN;

    const int n0 = wave * 32 + m;        // column of tile 2w for this lane
    const int n1 = n0 + 16;              // column of tile 2w+1
    const _Float16* bl0 = wlh + (size_t)n0 * D_IN;
    const _Float16* br0 = wrh + (size_t)n0 * D_IN;
    const _Float16* bl1 = wlh + (size_t)n1 * D_IN;
    const _Float16* br1 = wrh + (size_t)n1 * D_IN;

    v8f c00 = {};   // node tile 0 x column tile 2w
    v8f c01 = {};   // node tile 0 x column tile 2w+1
    v8f c10 = {};   // node tile 1 x column tile 2w
    v8f c11 = {};   // node tile 1 x column tile 2w+1
#pragma unroll
    for (int k0 = 0; k0 < D_IN; k0 += 32) {
        v16h aA0 = load_a16(arow0, k0, half);
        v16h aX0 = load_a16(xrow0, k0, half);
        v16h aA1 = load_a16(arow1, k0, half);
        v16h aX1 = load_a16(xrow1, k0, half);
        v16h wl0 = load_b16(bl0, k0, half);
        v16h wr0 = load_b16(br0, k0, half);
        v16h wl1 = load_b16(bl1, k0, half);
        v16h wr1 = load_b16(br1, k0, half);
        c00 = wmma_f16(aA0, wl0, c00);
        c00 = wmma_f16(aX0, wr0, c00);
        c01 = wmma_f16(aA0, wl1, c01);
        c01 = wmma_f16(aX0, wr1, c01);
        c10 = wmma_f16(aA1, wl0, c10);
        c10 = wmma_f16(aX1, wr0, c10);
        c11 = wmma_f16(aA1, wl1, c11);
        c11 = wmma_f16(aX1, wr1, c11);
    }

    // D layout: c[j] = h[nodeTileBase + j + 8*half, column n]; bias+relu then
    // sum this lane's node-rows (8 per tile, both tiles) -> column partial.
    const float b0  = bias2[n0];
    const float b1v = bias2[n1];
    float p0 = 0.0f, p1 = 0.0f;
#pragma unroll
    for (int j = 0; j < 8; ++j) {
        p0 += fmaxf(c00[j] + b0, 0.0f) + fmaxf(c10[j] + b0, 0.0f);
        p1 += fmaxf(c01[j] + b1v, 0.0f) + fmaxf(c11[j] + b1v, 0.0f);
    }
    __hip_atomic_fetch_add(&sPool[n0], p0, __ATOMIC_RELAXED, __HIP_MEMORY_SCOPE_WORKGROUP);
    __hip_atomic_fetch_add(&sPool[n1], p1, __ATOMIC_RELAXED, __HIP_MEMORY_SCOPE_WORKGROUP);
    __syncthreads();

    partials[(size_t)blockIdx.x * D_HID + tid] = sPool[tid];
}

// ---------------------------------------------------------------------------
// Stage 3: deterministic pooled reduction + MLP head + log_softmax.
// ---------------------------------------------------------------------------
__global__ __launch_bounds__(256)
void head_kernel(const float* __restrict__ partials, int nBlocks,
                 const float* __restrict__ W1, const float* __restrict__ b1,
                 const float* __restrict__ W2, const float* __restrict__ b2,
                 float* __restrict__ out)
{
    __shared__ float pooled[D_HID];
    __shared__ float z1[D_HID];
    __shared__ float zlog[N_CLS];
    const int n = threadIdx.x;

    float sum = 0.0f;
    for (int b = 0; b < nBlocks; ++b)
        sum += partials[(size_t)b * D_HID + n];
    pooled[n] = sum * (1.0f / (float)N_NODES);
    __syncthreads();

    float acc = b1[n];
    const float* w1row = W1 + (size_t)n * D_HID;
    for (int k = 0; k < D_HID; ++k) acc += w1row[k] * pooled[k];
    z1[n] = fmaxf(acc, 0.0f);
    __syncthreads();

    if (n < N_CLS) {
        float z = b2[n];
        const float* w2row = W2 + (size_t)n * D_HID;
        for (int k = 0; k < D_HID; ++k) z += w2row[k] * z1[k];
        zlog[n] = z;
    }
    __syncthreads();

    if (n == 0) {
        float mx = zlog[0];
        for (int i = 1; i < N_CLS; ++i) mx = fmaxf(mx, zlog[i]);
        float se = 0.0f;
        for (int i = 0; i < N_CLS; ++i) se += expf(zlog[i] - mx);
        float lse = mx + logf(se);
        for (int i = 0; i < N_CLS; ++i) out[i] = zlog[i] - lse;
    }
}

// ---------------------------------------------------------------------------
// kernel_launch
// Inputs (setup_inputs order): x, edge_index (int32: JAX x64 disabled),
// Wl[3,256,128], Wr[3,256,128], bias[3,256], W1, b1, W2, b2.
// Output: 10 f32 (log_softmax logits).
// ---------------------------------------------------------------------------
extern "C" void kernel_launch(void* const* d_in, const int* in_sizes, int n_in,
                              void* d_out, int out_size, void* d_ws, size_t ws_size,
                              hipStream_t stream) {
    const float* x    = (const float*)d_in[0];
    const int*   ei   = (const int*)  d_in[1];
    const float* Wl   = (const float*)d_in[2];
    const float* Wr   = (const float*)d_in[3];
    const float* bias = (const float*)d_in[4];
    const float* W1   = (const float*)d_in[5];
    const float* b1   = (const float*)d_in[6];
    const float* W2   = (const float*)d_in[7];
    const float* b2   = (const float*)d_in[8];
    float* out = (float*)d_out;

    const int nEdges = in_sizes[1] / 2;

    // Workspace layout (f32 units; all fp16 regions land on 32B boundaries):
    //   deg[100000] | partials[3125*256] | aggsum[100000*128]
    //   | xh (f16, 100000*128) | aggh (f16, 100000*128)
    //   | wlh (f16, 256*128)   | wrh (f16, 256*128)
    float* ws       = (float*)d_ws;
    float* deg      = ws;
    float* partials = deg + N_NODES;
    float* aggsum   = partials + (size_t)LAYER_BLOCKS * D_HID;
    float* fp32End  = aggsum + (size_t)N_NODES * D_IN;
    _Float16* xh    = (_Float16*)fp32End;
    _Float16* aggh  = xh   + (size_t)N_NODES * D_IN;
    _Float16* wlh   = aggh + (size_t)N_NODES * D_IN;
    _Float16* wrh   = wlh  + (size_t)D_HID * D_IN;

    const size_t zeroTotal =
        (size_t)N_NODES + (size_t)LAYER_BLOCKS * D_HID + (size_t)N_NODES * D_IN;

    // Only layer 2 of the reference contributes to the output.
    const float* Wl2   = Wl   + 2 * (size_t)D_HID * D_IN;
    const float* Wr2   = Wr   + 2 * (size_t)D_HID * D_IN;
    const float* bias2 = bias + 2 * (size_t)D_HID;

    const size_t nodes4 = (size_t)N_NODES * D_IN / 4;   // float4 groups
    const size_t w4     = (size_t)D_HID * D_IN / 4;

    zero_ws_kernel<<<(unsigned)((zeroTotal + 255) / 256), 256, 0, stream>>>(ws, zeroTotal);

    // Conversions independent of the edge pass:
    cvt_f16_kernel<<<(unsigned)((nodes4 + 255) / 256), 256, 0, stream>>>(x, xh, nodes4);
    cvt_f16_kernel<<<(unsigned)((w4 + 255) / 256), 256, 0, stream>>>(Wl2, wlh, w4);
    cvt_f16_kernel<<<(unsigned)((w4 + 255) / 256), 256, 0, stream>>>(Wr2, wrh, w4);

    edge_agg_kernel<<<(nEdges + 7) / 8, 256, 0, stream>>>(ei, x, aggsum, deg, nEdges);

    cvt_agg_kernel<<<(unsigned)((nodes4 + 255) / 256), 256, 0, stream>>>(aggsum, deg, aggh,
                                                                         nodes4);

    sage_layer_kernel<<<LAYER_BLOCKS, 256, 0, stream>>>(xh, aggh, wlh, wrh, bias2, partials);

    head_kernel<<<1, 256, 0, stream>>>(partials, LAYER_BLOCKS, W1, b1, W2, b2, out);
}
